// MultiHeadAttention_3066606649915
// MI455X (gfx1250) — matrix operand order repaired and checked by reading, not compiled, not run on hardware
//
#include <hip/hip_runtime.h>
#include <hip/hip_bf16.h>

// ---------------------------------------------------------------------------
// MHA forward for MI455X (gfx1250, wave32, WMMA 16x16x32 bf16 w/ f32 accum).
//   1) f32 -> bf16 conversion of X; W_qkv / W_out converted AND transposed
//      to [n][k] so every GEMM tile is a contiguous 2D tile.
//   2) GEMMs stage A and W tiles with the Tensor Data Mover
//      (tensor_load_to_lds, 2D descriptor, HW padding = LDS bank padding),
//      double-buffered, one issuing wave, s_wait_tensorcnt + barrier.
//   3) Flash attention: 16-row query tile per wave, online softmax in regs,
//      P relayout (C-frag -> A-frag) via per-wave LDS strip, V pre-transposed.
//   4) Output GEMM + bias -> f32 d_out.
// ---------------------------------------------------------------------------

typedef __attribute__((ext_vector_type(4)))  unsigned int u32x4;
typedef __attribute__((ext_vector_type(8)))  unsigned int u32x8;
typedef __attribute__((ext_vector_type(4)))  float        f32x4;
typedef __attribute__((ext_vector_type(8)))  float        v8f;
typedef __attribute__((ext_vector_type(16))) __bf16       v16bf;
typedef __attribute__((ext_vector_type(4)))  __bf16       bf16x4;

static constexpr int Bc  = 4;
static constexpr int Tc  = 2048;
static constexpr int Dc  = 1024;
static constexpr int NHc = 16;
static constexpr int HDc = 64;
static constexpr int N3c = 3 * Dc;
static constexpr int Mc  = Bc * Tc;      // 8192 rows

union Frag { v16bf v; u32x4 u[2]; };

// A-fragment (16x32 bf16): lane covers 8 contiguous K at `base`, next 8 at +16.
static __device__ __forceinline__ v16bf load_frag_a(const __bf16* base) {
  Frag f;
  f.u[0] = *(const u32x4*)(base);
  f.u[1] = *(const u32x4*)(base + 16);
  return f.v;
}
// B-fragment (32x16 bf16): lane covers 8 contiguous K at `base`, next 8 at +16.
static __device__ __forceinline__ v16bf load_frag_b(const __bf16* base) {
  Frag f;
  f.u[0] = *(const u32x4*)(base);
  f.u[1] = *(const u32x4*)(base + 16);
  return f.v;
}

static __device__ __forceinline__ v8f wmma_bf16(v16bf a, v16bf b, v8f c) {
  return __builtin_amdgcn_wmma_f32_16x16x32_bf16(false, a, false, b,
                                                 (short)0, c, false, false);
}

// ---------------------------------------------------------------------------
// TDM: load one 2D tile (tile_dim0 = 64 bf16 = 128B rows, tile_dim1 = 128 rows)
// from a row-major bf16 tensor (row stride = stride_elems) into LDS.
// HW padding: +4 DWORDs (8 elems) after every 32 DWORDs (64 elems) stored
// -> LDS row stride 72 elems, matching LDAS/LDBS below.
// Descriptor groups per CDNA5 ISA 8.3/8.4 (2-group form, tensors <= 2D).
// ---------------------------------------------------------------------------
static __device__ __forceinline__ void tdm_load_tile_2d(
    unsigned lds_byte, const void* gptr, unsigned stride_elems,
    unsigned tensor_d1) {
  unsigned long long ga = (unsigned long long)(uintptr_t)gptr;
  u32x4 g0;
  g0[0] = 1u;                                           // count=1, user desc
  g0[1] = lds_byte;                                     // lds_addr
  g0[2] = (unsigned)ga;                                 // global_addr[31:0]
  g0[3] = (unsigned)((ga >> 32) & 0x01ffffffull)        // global_addr[56:32]
        | (2u << 30);                                   // type=2 ("image")
  u32x8 g1;
  g1[0] = (1u << 16)                                    // data_size = 2B
        | (1u << 20)                                    // pad_enable
        | (4u << 22)                                    // pad_interval: 32 DW
        | (3u << 25);                                   // pad_amount: 4 DW
  g1[1] = (stride_elems & 0xffffu) << 16;               // tensor_dim0 lo16
  g1[2] = (stride_elems >> 16)                          // tensor_dim0 hi16
        | ((tensor_d1 & 0xffffu) << 16);                // tensor_dim1 lo16
  g1[3] = (tensor_d1 >> 16)                             // tensor_dim1 hi16
        | (64u << 16);                                  // tile_dim0 = 64
  g1[4] = 128u;                                         // tile_dim1 = 128
  g1[5] = stride_elems;                                 // dim0_stride[31:0]
  g1[6] = 0u;                                           // dim0_stride hi, d1s lo
  g1[7] = 0u;
  asm volatile("tensor_load_to_lds %0, %1" :: "s"(g0), "s"(g1) : "memory");
}

// ---------------------------------------------------------------------------
// f32 -> bf16 conversion, 4-wide
// ---------------------------------------------------------------------------
__global__ __launch_bounds__(256) void cvt_f32_to_bf16(
    const float* __restrict__ in, __bf16* __restrict__ out, int n4) {
  int i = blockIdx.x * blockDim.x + threadIdx.x;
  if (i < n4) {
    f32x4 v = ((const f32x4*)in)[i];
    bf16x4 o;
    o[0] = (__bf16)v[0]; o[1] = (__bf16)v[1];
    o[2] = (__bf16)v[2]; o[3] = (__bf16)v[3];
    ((bf16x4*)out)[i] = o;
  }
}

// f32 [K][N] -> bf16 [N][K] (weight pre-transpose; coalesced reads, 16B writes)
__global__ __launch_bounds__(256) void cvt_transpose_bf16(
    const float* __restrict__ in, __bf16* __restrict__ out, int K, int N) {
  int n  = blockIdx.x * 256 + threadIdx.x;
  int k0 = blockIdx.y * 8;
  union { u32x4 u; __bf16 h[8]; } v;
#pragma unroll
  for (int j = 0; j < 8; ++j)
    v.h[j] = (__bf16)in[(size_t)(k0 + j) * N + n];
  *(u32x4*)&out[(size_t)n * K + k0] = v.u;
}

// ---------------------------------------------------------------------------
// bf16 GEMM with TDM-staged, double-buffered tiles.
// C[M,N] = A[M,K] @ Wt[N,K]^T + bias.  Block tile 128x128, K-step 64.
// 8 waves as 4(M) x 2(N); wave tile 32x64 -> 2x4 C fragments.
// EPI 0: scatter to Q (b,h,t,e), K (b,h,t,e), Vt (b,h,e,t)  [bf16]
// EPI 1: plain f32 store
// ---------------------------------------------------------------------------
static constexpr int BM = 128, BN = 128, BK = 64;
static constexpr int LDAS = BK + 8;   // 72 elems = TDM-padded row stride
static constexpr int LDBS = BK + 8;

template <int EPI>
__global__ __launch_bounds__(256) void gemm_bf16_tdm_kernel(
    const __bf16* __restrict__ A, const __bf16* __restrict__ Wt,
    const float* __restrict__ bias,
    int Mdim, int Ndim, int Kdim,
    __bf16* __restrict__ Qout, __bf16* __restrict__ Kout,
    __bf16* __restrict__ Vtout, float* __restrict__ Cout) {
  const int tid   = threadIdx.x;
  const int wave  = tid >> 5, lane = tid & 31;
  const int wm    = wave >> 1, wn = wave & 1;
  const int laneM = lane & 15, laneH = lane >> 4;
  const int m0    = blockIdx.y * BM;
  const int n0    = blockIdx.x * BN;

  __shared__ __bf16 As[2][BM * LDAS];   // [m][k], k contiguous
  __shared__ __bf16 Bs[2][BN * LDBS];   // [n][k], k contiguous

  v8f acc[2][4] = {};

  if (wave == 0) {
    tdm_load_tile_2d((unsigned)(uintptr_t)(void*)&As[0][0],
                     &A[(size_t)m0 * Kdim], (unsigned)Kdim, 128u);
    tdm_load_tile_2d((unsigned)(uintptr_t)(void*)&Bs[0][0],
                     &Wt[(size_t)n0 * Kdim], (unsigned)Kdim, 128u);
  }

  int par = 0;
  for (int kt = 0; kt < Kdim; kt += BK) {
    const bool more = (kt + BK) < Kdim;
    if (wave == 0) {
      if (more) {
        tdm_load_tile_2d((unsigned)(uintptr_t)(void*)&As[par ^ 1][0],
                         &A[(size_t)m0 * Kdim + kt + BK], (unsigned)Kdim, 128u);
        tdm_load_tile_2d((unsigned)(uintptr_t)(void*)&Bs[par ^ 1][0],
                         &Wt[(size_t)n0 * Kdim + kt + BK], (unsigned)Kdim, 128u);
        __builtin_amdgcn_s_wait_tensorcnt(2);   // prev batch done (in-order)
      } else {
        __builtin_amdgcn_s_wait_tensorcnt(0);
      }
    }
    __syncthreads();

    const __bf16* as = &As[par][0];
    const __bf16* bs = &Bs[par][0];
#pragma unroll
    for (int kc = 0; kc < 2; ++kc) {
      v16bf af[2];
#pragma unroll
      for (int mf = 0; mf < 2; ++mf)
        af[mf] = load_frag_a(&as[(wm * 32 + mf * 16 + laneM) * LDAS +
                                 kc * 32 + laneH * 8]);
#pragma unroll
      for (int nf = 0; nf < 4; ++nf) {
        v16bf bf = load_frag_b(&bs[(wn * 64 + nf * 16 + laneM) * LDBS +
                                   kc * 32 + laneH * 8]);
        acc[0][nf] = wmma_bf16(af[0], bf, acc[0][nf]);
        acc[1][nf] = wmma_bf16(af[1], bf, acc[1][nf]);
      }
    }
    __syncthreads();
    par ^= 1;
  }

  // ---- epilogue ----
#pragma unroll
  for (int mf = 0; mf < 2; ++mf) {
#pragma unroll
    for (int nf = 0; nf < 4; ++nf) {
#pragma unroll
      for (int r = 0; r < 8; ++r) {
        int row = m0 + wm * 32 + mf * 16 + laneH * 8 + r;
        int col = n0 + wn * 64 + nf * 16 + laneM;
        float v = acc[mf][nf][r] + bias[col];
        if (EPI == 0) {
          int h   = col / (3 * HDc);
          int rem = col % (3 * HDc);
          int sel = rem / HDc;
          int e   = rem % HDc;
          int b   = row / Tc;
          int t   = row % Tc;
          int bh  = b * NHc + h;
          if (sel == 0)
            Qout[((size_t)bh * Tc + t) * HDc + e] = (__bf16)v;
          else if (sel == 1)
            Kout[((size_t)bh * Tc + t) * HDc + e] = (__bf16)v;
          else
            Vtout[((size_t)bh * HDc + e) * Tc + t] = (__bf16)v;
        } else {
          Cout[(size_t)row * Ndim + col] = v;
        }
      }
    }
  }
}

// ---------------------------------------------------------------------------
// Flash attention (causal).  grid = B*NH*(T/128); 8 waves, 16 q-rows per wave.
// ---------------------------------------------------------------------------
__global__ __launch_bounds__(256) void flash_attn_kernel(
    const __bf16* __restrict__ Qb, const __bf16* __restrict__ Kb,
    const __bf16* __restrict__ Vt, __bf16* __restrict__ Ob) {
  const int tid   = threadIdx.x;
  const int wave  = tid >> 5, lane = tid & 31;
  const int laneM = lane & 15, laneH = lane >> 4;
  const int gid   = blockIdx.x;
  const int bh    = gid >> 4;      // T/128 == 16 query blocks
  const int qblk  = gid & 15;
  const int q0    = qblk * 128 + wave * 16;
  const int bIdx  = bh >> 4;       // / NH
  const int h     = bh & 15;

  __shared__ __bf16 Ps[8 * 16 * 72];
  __bf16* ps = &Ps[wave * 16 * 72];

  const __bf16* Qh = Qb + (size_t)bh * Tc * HDc;
  const __bf16* Kh = Kb + (size_t)bh * Tc * HDc;
  const __bf16* Vh = Vt + (size_t)bh * HDc * Tc;   // (e, t), t contiguous

  v16bf qf[2];
#pragma unroll
  for (int c = 0; c < 2; ++c)
    qf[c] = load_frag_a(&Qh[(size_t)(q0 + laneM) * HDc + c * 32 + laneH * 8]);

  v8f   o[4] = {};
  float mrow[8], lrow[8];
#pragma unroll
  for (int r = 0; r < 8; ++r) { mrow[r] = -1e30f; lrow[r] = 0.f; }

  for (int kb = 0; kb < q0 + 16; kb += 64) {
    if (kb + 64 < Tc) {           // hint next K/V block into the caches
      __builtin_prefetch(&Kh[(size_t)(kb + 64) * HDc], 0, 1);
      __builtin_prefetch(&Vh[kb + 64], 0, 1);
    }
    // ---- S = Q @ K^T (16 queries x 64 keys) ----
    v8f s[4] = {};
#pragma unroll
    for (int kc = 0; kc < 2; ++kc) {
#pragma unroll
      for (int nf = 0; nf < 4; ++nf) {
        v16bf kfr = load_frag_b(
            &Kh[(size_t)(kb + nf * 16 + laneM) * HDc + kc * 32 + laneH * 8]);
        s[nf] = wmma_bf16(qf[kc], kfr, s[nf]);
      }
    }
    // ---- scale + causal mask + row max ----
    float mb[8];
#pragma unroll
    for (int r = 0; r < 8; ++r) mb[r] = -1e30f;
#pragma unroll
    for (int nf = 0; nf < 4; ++nf) {
      int colk = kb + nf * 16 + laneM;
#pragma unroll
      for (int r = 0; r < 8; ++r) {
        int rowq = q0 + laneH * 8 + r;
        float sv = s[nf][r] * 0.125f;            // 1/sqrt(64)
        sv       = (colk > rowq) ? -1e30f : sv;
        s[nf][r] = sv;
        mb[r]    = fmaxf(mb[r], sv);
      }
    }
#pragma unroll
    for (int r = 0; r < 8; ++r) {
#pragma unroll
      for (int msk = 1; msk < 16; msk <<= 1)
        mb[r] = fmaxf(mb[r], __shfl_xor(mb[r], msk));
    }
    float alpha[8];
#pragma unroll
    for (int r = 0; r < 8; ++r) {
      float mn = fmaxf(mrow[r], mb[r]);
      alpha[r] = __expf(mrow[r] - mn);
      mrow[r]  = mn;
    }
    // ---- P = exp(S - m), row sums, rescale O ----
    float lb[8];
#pragma unroll
    for (int r = 0; r < 8; ++r) lb[r] = 0.f;
#pragma unroll
    for (int nf = 0; nf < 4; ++nf) {
#pragma unroll
      for (int r = 0; r < 8; ++r) {
        float p  = __expf(s[nf][r] - mrow[r]);
        s[nf][r] = p;
        lb[r]   += p;
      }
    }
#pragma unroll
    for (int r = 0; r < 8; ++r) {
#pragma unroll
      for (int msk = 1; msk < 16; msk <<= 1)
        lb[r] += __shfl_xor(lb[r], msk);
      lrow[r] = lrow[r] * alpha[r] + lb[r];
    }
#pragma unroll
    for (int nf = 0; nf < 4; ++nf)
#pragma unroll
      for (int r = 0; r < 8; ++r) o[nf][r] *= alpha[r];

    // ---- C-layout -> A-layout via per-wave LDS strip (bf16) ----
#pragma unroll
    for (int nf = 0; nf < 4; ++nf)
#pragma unroll
      for (int r = 0; r < 8; ++r)
        ps[(laneH * 8 + r) * 72 + nf * 16 + laneM] = (__bf16)s[nf][r];

    // ---- O += P @ V ----
#pragma unroll
    for (int kc = 0; kc < 2; ++kc) {
      v16bf pf = load_frag_a(&ps[laneM * 72 + kc * 32 + laneH * 8]);
#pragma unroll
      for (int nf = 0; nf < 4; ++nf) {
        v16bf vf = load_frag_b(
            &Vh[(size_t)(nf * 16 + laneM) * Tc + kb + kc * 32 + laneH * 8]);
        o[nf] = wmma_bf16(pf, vf, o[nf]);
      }
    }
  }

  // ---- normalize + store (B, T, D) bf16 ----
#pragma unroll
  for (int r = 0; r < 8; ++r) lrow[r] = 1.f / lrow[r];
#pragma unroll
  for (int nf = 0; nf < 4; ++nf) {
#pragma unroll
    for (int r = 0; r < 8; ++r) {
      int q = q0 + laneH * 8 + r;
      int e = nf * 16 + laneM;
      Ob[((size_t)bIdx * Tc + q) * Dc + h * HDc + e] =
          (__bf16)(o[nf][r] * lrow[r]);
    }
  }
}

// ---------------------------------------------------------------------------
extern "C" void kernel_launch(void* const* d_in, const int* in_sizes, int n_in,
                              void* d_out, int out_size, void* d_ws,
                              size_t ws_size, hipStream_t stream) {
  (void)in_sizes; (void)n_in; (void)out_size; (void)ws_size;
  const float* X    = (const float*)d_in[0];
  const float* Wqkv = (const float*)d_in[1];
  const float* bqkv = (const float*)d_in[2];
  const float* Wout = (const float*)d_in[3];
  const float* bout = (const float*)d_in[4];
  float* out = (float*)d_out;

  // Workspace carve-up (~92.3 MB total)
  char* ws = (char*)d_ws;
  __bf16* Xb  = (__bf16*)ws; ws += (size_t)Mc * Dc * 2;
  __bf16* Wqt = (__bf16*)ws; ws += (size_t)Dc * N3c * 2;  // [3072][1024]
  __bf16* Wot = (__bf16*)ws; ws += (size_t)Dc * Dc * 2;   // [1024][1024]
  __bf16* Qb  = (__bf16*)ws; ws += (size_t)Mc * Dc * 2;
  __bf16* Kb  = (__bf16*)ws; ws += (size_t)Mc * Dc * 2;
  __bf16* Vt  = (__bf16*)ws; ws += (size_t)Mc * Dc * 2;
  __bf16* Ab  = (__bf16*)ws; ws += (size_t)Mc * Dc * 2;

  {
    int n4 = (Mc * Dc) / 4;
    cvt_f32_to_bf16<<<n4 / 256, 256, 0, stream>>>(X, Xb, n4);
  }
  // W_qkv [1024][3072] -> Wqt [3072][1024]
  cvt_transpose_bf16<<<dim3(N3c / 256, Dc / 8), 256, 0, stream>>>(
      Wqkv, Wqt, Dc, N3c);
  // W_out [1024][1024] -> Wot [1024][1024]
  cvt_transpose_bf16<<<dim3(Dc / 256, Dc / 8), 256, 0, stream>>>(
      Wout, Wot, Dc, Dc);

  // QKV projection: 8192x3072 = 64 x 24 tiles
  gemm_bf16_tdm_kernel<0><<<dim3(N3c / BN, Mc / BM), 256, 0, stream>>>(
      Xb, Wqt, bqkv, Mc, N3c, Dc, Qb, Kb, Vt, nullptr);

  // Flash attention: 64 (b,h) x 16 query blocks
  flash_attn_kernel<<<Bc * NHc * (Tc / 128), 256, 0, stream>>>(Qb, Kb, Vt, Ab);

  // Output projection: 8192x1024 = 64 x 8 tiles, f32 out
  gemm_bf16_tdm_kernel<1><<<dim3(Dc / BN, Mc / BM), 256, 0, stream>>>(
      Ab, Wot, bout, Mc, Dc, Dc, nullptr, nullptr, nullptr, out);
}
